// MessagePassingElectionModel_6571299962912
// MI455X (gfx1250) — compile-verified, matrix operand order checked by reading, not run on hardware
//
#include <hip/hip_runtime.h>
#include <hip/hip_bf16.h>

typedef __attribute__((ext_vector_type(16))) _Float16 v16h;
typedef __attribute__((ext_vector_type(8)))  float    v8f;

#define NN    100000
#define NE    1600000
#define NCAND 10000
#define NG    1000
#define ND    32
#define ED    8
#define NL    4

#define EFRAGS (NL * 3 * 32)        // edge B fragments: 4 layers x 3 K-chunks x 32 lanes
#define NFRAGS (NL * 4 * 32)        // node B fragments: 4 layers x (2 K-chunks x 2 col-tiles) x 32 lanes

// ---------------- input embeddings ----------------
__global__ __launch_bounds__(256) void k_node_in(const float* __restrict__ x,
    const float* __restrict__ W, const float* __restrict__ b, float* __restrict__ h)
{
  int t = blockIdx.x * 256 + threadIdx.x;
  if (t >= NN * ND) return;
  int n = t >> 5, d = t & 31;
  h[t] = x[2 * n] * W[d] + x[2 * n + 1] * W[ND + d] + b[d];
}

__global__ __launch_bounds__(256) void k_edge_in(const float* __restrict__ ea,
    const float* __restrict__ W, const float* __restrict__ b, float* __restrict__ e)
{
  int t = blockIdx.x * 256 + threadIdx.x;
  if (t >= NE * ED) return;
  int i = t >> 3, d = t & 7;
  e[t] = ea[i] * W[d] + b[d];
}

__global__ __launch_bounds__(256) void k_zero(float* __restrict__ p, int n)
{
  int t = blockIdx.x * 256 + threadIdx.x;
  if (t < n) p[t] = 0.f;
}

// ---------------- B-fragment pre-swizzle (runs once, cold) ----------------
__global__ __launch_bounds__(256) void k_prep_frags(
    const float* __restrict__ We_l, const float* __restrict__ Wn_l,
    _Float16* __restrict__ ebfrag, _Float16* __restrict__ nbfrag)
{
  int t = blockIdx.x * 256 + threadIdx.x;
  if (t < EFRAGS) {
    const int lane = t & 31;
    const int c    = (t >> 5) % 3;
    const int l    = t / (3 * 32);
    const int n    = lane & 15;
    const int khalf = (lane >> 4) << 3;
    _Float16* o = ebfrag + (size_t)t * 16;
    #pragma unroll
    for (int i = 0; i < 16; ++i) {
      const int r = i >> 1, j = i & 1;
      const int k = c * 32 + ((r & 3) << 1) + ((r >> 2) << 4) + khalf + j;
      float w = (k < 2 * ND + ED && n < ED)
                  ? We_l[((size_t)l * (2 * ND + ED) + k) * ED + n] : 0.f;
      o[i] = (_Float16)w;
    }
  } else if (t < EFRAGS + NFRAGS) {
    const int u    = t - EFRAGS;
    const int lane = u & 31;
    const int ct   = (u >> 5) & 3;     // c*2 + tile
    const int l    = u >> 7;
    const int c    = ct >> 1;
    const int tt   = ct & 1;
    const int n    = (lane & 15) + (tt << 4);
    const int khalf = (lane >> 4) << 3;
    _Float16* o = nbfrag + (size_t)u * 16;
    #pragma unroll
    for (int i = 0; i < 16; ++i) {
      const int r = i >> 1, j = i & 1;
      const int k = c * 32 + ((r & 3) << 1) + ((r >> 2) << 4) + khalf + j;
      float w = (k < ND + ED) ? Wn_l[((size_t)l * (ND + ED) + k) * ND + n] : 0.f;
      o[i] = (_Float16)w;
    }
  }
}

// pack two contiguous 8-float segments into one v16h A fragment
__device__ __forceinline__ v16h pack16(float4 a0, float4 a1, float4 b0, float4 b1)
{
  v16h v;
  v[0]  = (_Float16)a0.x; v[1]  = (_Float16)a0.y;
  v[2]  = (_Float16)a0.z; v[3]  = (_Float16)a0.w;
  v[4]  = (_Float16)a1.x; v[5]  = (_Float16)a1.y;
  v[6]  = (_Float16)a1.z; v[7]  = (_Float16)a1.w;
  v[8]  = (_Float16)b0.x; v[9]  = (_Float16)b0.y;
  v[10] = (_Float16)b0.z; v[11] = (_Float16)b0.w;
  v[12] = (_Float16)b1.x; v[13] = (_Float16)b1.y;
  v[14] = (_Float16)b1.z; v[15] = (_Float16)b1.w;
  return v;
}

// ---------------- edge MLP + scatter-add (WMMA) ----------------
__global__ __launch_bounds__(256) void k_edge_update(
    const float* __restrict__ h, float* __restrict__ e, float* __restrict__ agg,
    const _Float16* __restrict__ Bf,   // this layer's 3x32 v16h fragments
    const float* __restrict__ be,
    const int* __restrict__ src, const int* __restrict__ dst)
{
  __shared__ float s_ne[8][16][20];           // [wave][row][col(+pad)]
  const int lane = threadIdx.x & 31;
  const int wv   = threadIdx.x >> 5;
  const int tile = blockIdx.x * 8 + wv;
  const int e0 = tile << 4;
  if (e0 >= NE) return;                       // wave-uniform: EXEC stays all-1s

  const int M     = lane & 15;
  const int khalf = (lane >> 4) << 3;
  const int eid   = e0 + M;
  const int sn    = src[eid];
  const int dn    = dst[eid];

  // ---- A fragments: branch-free vector loads ----
  const float* hs = h + sn * ND + khalf;
  v16h a0 = pack16(*(const float4*)(hs),      *(const float4*)(hs + 4),
                   *(const float4*)(hs + 16), *(const float4*)(hs + 20));
  const float* hd = h + dn * ND + khalf;
  v16h a1 = pack16(*(const float4*)(hd),      *(const float4*)(hd + 4),
                   *(const float4*)(hd + 16), *(const float4*)(hd + 20));

  const float* ep = e + (size_t)eid * ED;     // all lanes load (shared cachelines)
  float4 er0 = *(const float4*)(ep);
  float4 er1 = *(const float4*)(ep + 4);
  const bool lo = lane < 16;                  // k=64..71 valid only for low K-half
  v16h a2 = {};
  a2[0] = lo ? (_Float16)er0.x : (_Float16)0.f;
  a2[1] = lo ? (_Float16)er0.y : (_Float16)0.f;
  a2[2] = lo ? (_Float16)er0.z : (_Float16)0.f;
  a2[3] = lo ? (_Float16)er0.w : (_Float16)0.f;
  a2[4] = lo ? (_Float16)er1.x : (_Float16)0.f;
  a2[5] = lo ? (_Float16)er1.y : (_Float16)0.f;
  a2[6] = lo ? (_Float16)er1.z : (_Float16)0.f;
  a2[7] = lo ? (_Float16)er1.w : (_Float16)0.f;

  // ---- B fragments: pre-swizzled, two b128 loads each ----
  const v16h* B = (const v16h*)Bf;
  v16h b0 = B[0 * 32 + lane];
  v16h b1 = B[1 * 32 + lane];
  v16h b2 = B[2 * 32 + lane];

  v8f acc = {};
  acc = __builtin_amdgcn_wmma_f32_16x16x32_f16(false, a0, false, b0, (short)0, acc, false, false);
  acc = __builtin_amdgcn_wmma_f32_16x16x32_f16(false, a1, false, b1, (short)0, acc, false, false);
  acc = __builtin_amdgcn_wmma_f32_16x16x32_f16(false, a2, false, b2, (short)0, acc, false, false);

  // ---- epilogue ----
  const int col     = lane & 15;
  const int rowbase = (lane >> 4) << 3;

  // share dst row ids cross-lane while EXEC is full (src lanes 0..15 all active)
  int drow[8];
  #pragma unroll
  for (int v = 0; v < 8; ++v)
    drow[v] = __builtin_amdgcn_ds_bpermute((rowbase + v) << 2, dn);

  // bias + ReLU; transpose new_e through LDS (all lanes write: cols 8..15 are
  // harmless scratch, never read back)
  const float bias = be[col & 7];
  float ne[8];
  #pragma unroll
  for (int v = 0; v < 8; ++v) {
    float t = acc[v] + bias;
    ne[v] = t > 0.f ? t : 0.f;
    s_ne[wv][rowbase + v][col] = ne[v];
  }

  // vectorized residual store: lane M (<16) still holds its old e row
  if (lane < 16) {
    const float* r = &s_ne[wv][lane][0];
    float4 n0 = *(const float4*)(r);
    float4 n1 = *(const float4*)(r + 4);
    float4 o0, o1;
    o0.x = er0.x + n0.x; o0.y = er0.y + n0.y;
    o0.z = er0.z + n0.z; o0.w = er0.w + n0.w;
    o1.x = er1.x + n1.x; o1.y = er1.y + n1.y;
    o1.z = er1.z + n1.z; o1.w = er1.w + n1.w;
    float4* eo = (float4*)(e + (size_t)eid * ED);
    eo[0] = o0;
    eo[1] = o1;
  }

  // atomic scatter-add into agg[dst] (irreducible segment_sum cost)
  if (col < ED) {
    #pragma unroll
    for (int v = 0; v < 8; ++v)
      atomicAdd(&agg[(size_t)drow[v] * ED + col], ne[v]);
  }
}

// ---------------- node MLP (WMMA) ----------------
__global__ __launch_bounds__(256) void k_node_update(
    float* __restrict__ h, const float* __restrict__ agg,
    const _Float16* __restrict__ Bf,   // this layer's 4x32 v16h fragments
    const float* __restrict__ bn)
{
  const int lane = threadIdx.x & 31;
  const int tile = blockIdx.x * 8 + (threadIdx.x >> 5);
  const int n0 = tile << 4;
  if (n0 >= NN) return;                       // wave-uniform

  const int M     = lane & 15;
  const int khalf = (lane >> 4) << 3;
  const int node  = n0 + M;

  const float* hp = h + node * ND + khalf;
  v16h a0 = pack16(*(const float4*)(hp),      *(const float4*)(hp + 4),
                   *(const float4*)(hp + 16), *(const float4*)(hp + 20));

  const float* ap = agg + (size_t)node * ED;
  float4 ag0 = *(const float4*)(ap);
  float4 ag1 = *(const float4*)(ap + 4);
  const bool lo = lane < 16;                  // k=32..39 valid only for low K-half
  v16h a1 = {};
  a1[0] = lo ? (_Float16)ag0.x : (_Float16)0.f;
  a1[1] = lo ? (_Float16)ag0.y : (_Float16)0.f;
  a1[2] = lo ? (_Float16)ag0.z : (_Float16)0.f;
  a1[3] = lo ? (_Float16)ag0.w : (_Float16)0.f;
  a1[4] = lo ? (_Float16)ag1.x : (_Float16)0.f;
  a1[5] = lo ? (_Float16)ag1.y : (_Float16)0.f;
  a1[6] = lo ? (_Float16)ag1.z : (_Float16)0.f;
  a1[7] = lo ? (_Float16)ag1.w : (_Float16)0.f;

  const v16h* B = (const v16h*)Bf;            // index (c*2 + tile)*32 + lane
  v8f acc0 = {}, acc1 = {};
  acc0 = __builtin_amdgcn_wmma_f32_16x16x32_f16(false, a0, false, B[0 * 32 + lane], (short)0, acc0, false, false);
  acc1 = __builtin_amdgcn_wmma_f32_16x16x32_f16(false, a0, false, B[1 * 32 + lane], (short)0, acc1, false, false);
  acc0 = __builtin_amdgcn_wmma_f32_16x16x32_f16(false, a1, false, B[2 * 32 + lane], (short)0, acc0, false, false);
  acc1 = __builtin_amdgcn_wmma_f32_16x16x32_f16(false, a1, false, B[3 * 32 + lane], (short)0, acc1, false, false);

  const int rowbase = (lane >> 4) << 3;
  const int col0 = lane & 15;
  const float bias0 = bn[col0];
  const float bias1 = bn[col0 + 16];
  #pragma unroll
  for (int v = 0; v < 8; ++v) {
    const int nd = n0 + rowbase + v;
    float nh0 = acc0[v] + bias0;
    float nh1 = acc1[v] + bias1;
    nh0 = nh0 > 0.f ? nh0 : 0.f;
    nh1 = nh1 > 0.f ? nh1 : 0.f;
    h[nd * ND + col0]      += nh0;            // residual, 64B-coalesced
    h[nd * ND + col0 + 16] += nh1;
  }
}

// ---------------- head: logits + segment log-softmax ----------------
__global__ __launch_bounds__(256) void k_logits(const float* __restrict__ h,
    const float* __restrict__ Wout, const float* __restrict__ bout,
    const int* __restrict__ cand, const int* __restrict__ batch,
    float* __restrict__ logits, int* __restrict__ seg)
{
  int c = blockIdx.x * 256 + threadIdx.x;
  if (c >= NCAND) return;
  int node = cand[c];
  float s = bout[0];
  const float4* hv = (const float4*)(h + node * ND);
  const float4* wv = (const float4*)Wout;
  #pragma unroll
  for (int q = 0; q < 8; ++q) {
    float4 hh = hv[q], ww = wv[q];
    s += hh.x * ww.x + hh.y * ww.y + hh.z * ww.z + hh.w * ww.w;
  }
  logits[c] = s;
  seg[c] = batch[node];
}

__global__ __launch_bounds__(256) void k_ginit(float* __restrict__ mx,
                                               float* __restrict__ sums)
{
  int g = blockIdx.x * 256 + threadIdx.x;
  if (g >= NG) return;
  mx[g] = -INFINITY;
  sums[g] = 0.f;
}

__device__ __forceinline__ void atomicMaxF(float* addr, float val)
{
  int* ia = (int*)addr;
  int old = __float_as_int(*addr);
  while (__int_as_float(old) < val) {
    int assumed = old;
    old = atomicCAS(ia, assumed, __float_as_int(val));
    if (old == assumed) break;
  }
}

__global__ __launch_bounds__(256) void k_gmax(const float* __restrict__ logits,
    const int* __restrict__ seg, float* __restrict__ mx)
{
  int c = blockIdx.x * 256 + threadIdx.x;
  if (c >= NCAND) return;
  atomicMaxF(&mx[seg[c]], logits[c]);
}

__global__ __launch_bounds__(256) void k_gsum(const float* __restrict__ logits,
    const int* __restrict__ seg, const float* __restrict__ mx,
    float* __restrict__ sums, float* __restrict__ shifted)
{
  int c = blockIdx.x * 256 + threadIdx.x;
  if (c >= NCAND) return;
  float sh = logits[c] - mx[seg[c]];
  shifted[c] = sh;
  atomicAdd(&sums[seg[c]], __expf(sh));
}

__global__ __launch_bounds__(256) void k_gout(const float* __restrict__ shifted,
    const int* __restrict__ seg, const float* __restrict__ sums,
    float* __restrict__ out)
{
  int c = blockIdx.x * 256 + threadIdx.x;
  if (c >= NCAND) return;
  out[c] = shifted[c] - __logf(sums[seg[c]]);
}

// ---------------- orchestration ----------------
extern "C" void kernel_launch(void* const* d_in, const int* in_sizes, int n_in,
                              void* d_out, int out_size, void* d_ws, size_t ws_size,
                              hipStream_t stream)
{
  const float* x     = (const float*)d_in[0];
  const float* ea    = (const float*)d_in[1];
  const float* Wn_in = (const float*)d_in[2];
  const float* bn_in = (const float*)d_in[3];
  const float* We_in = (const float*)d_in[4];
  const float* be_in = (const float*)d_in[5];
  const float* We_l  = (const float*)d_in[6];
  const float* be_l  = (const float*)d_in[7];
  const float* Wn_l  = (const float*)d_in[8];
  const float* bn_l  = (const float*)d_in[9];
  const float* Wout  = (const float*)d_in[10];
  const float* bout  = (const float*)d_in[11];
  const int*   eidx  = (const int*)d_in[12];
  const int*   batch = (const int*)d_in[13];
  const int*   cand  = (const int*)d_in[14];
  float* out = (float*)d_out;

  // workspace carve-out
  float* h       = (float*)d_ws;
  float* e       = h + (size_t)NN * ND;
  float* agg     = e + (size_t)NE * ED;
  float* logits  = agg + (size_t)NN * ED;
  float* shifted = logits + NCAND;
  float* mx      = shifted + NCAND;
  float* sums    = mx + NG;
  int*   seg     = (int*)(sums + NG);
  size_t fragoff = (((size_t)(seg + NCAND) - (size_t)d_ws) + 31) & ~(size_t)31;
  _Float16* ebfrag = (_Float16*)((char*)d_ws + fragoff);           // EFRAGS*16 halves
  _Float16* nbfrag = ebfrag + (size_t)EFRAGS * 16;                 // NFRAGS*16 halves

  const int* src = eidx;
  const int* dst = eidx + NE;

  k_prep_frags<<<(EFRAGS + NFRAGS + 255) / 256, 256, 0, stream>>>(We_l, Wn_l,
                                                                  ebfrag, nbfrag);
  k_node_in<<<(NN * ND + 255) / 256, 256, 0, stream>>>(x, Wn_in, bn_in, h);
  k_edge_in<<<(NE * ED + 255) / 256, 256, 0, stream>>>(ea, We_in, be_in, e);

  for (int l = 0; l < NL; ++l) {
    k_zero<<<(NN * ED + 255) / 256, 256, 0, stream>>>(agg, NN * ED);
    k_edge_update<<<NE / 128, 256, 0, stream>>>(
        h, e, agg, ebfrag + (size_t)l * 3 * 32 * 16, be_l + l * ED, src, dst);
    k_node_update<<<(NN / 16 + 7) / 8, 256, 0, stream>>>(
        h, agg, nbfrag + (size_t)l * 4 * 32 * 16, bn_l + l * ND);
  }

  k_logits<<<(NCAND + 255) / 256, 256, 0, stream>>>(h, Wout, bout, cand, batch,
                                                    logits, seg);
  k_ginit<<<(NG + 255) / 256, 256, 0, stream>>>(mx, sums);
  k_gmax<<<(NCAND + 255) / 256, 256, 0, stream>>>(logits, seg, mx);
  k_gsum<<<(NCAND + 255) / 256, 256, 0, stream>>>(logits, seg, mx, sums, shifted);
  k_gout<<<(NCAND + 255) / 256, 256, 0, stream>>>(shifted, seg, sums, out);
}
